// Base_GNN_Model_57767310131303
// MI455X (gfx1250) — matrix operand order repaired and checked by reading, not compiled, not run on hardware
//
#include <hip/hip_runtime.h>
#include <hip/hip_bf16.h>

// ---------------------------------------------------------------------------
// GNN pipeline for MI455X (gfx1250, wave32, WMMA).
// All GEMMs run on v_wmma_f32_16x16x32_bf16 (bf16 in, f32 accumulate).
// Weights are pre-packed per launch into the CDNA5 B-fragment lane layout so
// each lane loads its whole fragment as one contiguous 32B chunk (L2-resident).
// Activations are converted f32->bf16 (RNE) into LDS in the A-fragment layout
// with a padded row stride (K/2+4 dwords) so the two ds_load_b128 per fragment
// are bank-conflict free across the 16 rows.
// Scatter phase uses hardware fp32 global atomics (global_atomic_add_f32).
// ---------------------------------------------------------------------------

#define NUSERS   100000
#define NPRODS   50000
#define NNODES   150000
#define NEDGES   250000

typedef __bf16 bf16x8  __attribute__((ext_vector_type(8)));
typedef __bf16 bf16x16 __attribute__((ext_vector_type(16)));
typedef float  f32x8   __attribute__((ext_vector_type(8)));

__device__ __forceinline__ unsigned pack2bf16(float a, float b) {
  // round-to-nearest-even f32 -> bf16, packed {b,a}
  unsigned ua = __float_as_uint(a), ub = __float_as_uint(b);
  ua += 0x7FFFu + ((ua >> 16) & 1u);
  ub += 0x7FFFu + ((ub >> 16) & 1u);
  return (ua >> 16) | (ub & 0xFFFF0000u);
}

__device__ __forceinline__ void atomAddF(float* p, float v) {
  unsafeAtomicAdd(p, v);   // -> global_atomic_add_f32 (no CAS loop)
}

// ---------------------------------------------------------------------------
// Pack f32 weight W[K][N] (row-major) into bf16 B-fragments for
// v_wmma_f32_16x16x32_bf16. Tile (kt,nt) covers K=kt*32..+31, N=nt*16..+15.
// Lane l supplies column n = nt*16 + (l&15); its 8 dwords are K pairs
// ((v>>2)*16 + khalf*8 + 2*(v&3), +1) with khalf = l>>4  [CDNA5 ISA 7.12.2 B layout].
// Memory layout: dword index = ((kt*ntn + nt)*32 + lane)*8 + v  (fully coalesced).
// ---------------------------------------------------------------------------
__global__ void k_pack_w(const float* __restrict__ W, unsigned* __restrict__ Wp,
                         int K, int N) {
  int tid = blockIdx.x * 256 + threadIdx.x;
  if (tid >= (K * N) / 2) return;
  int v       = tid & 7;
  int lane    = (tid >> 3) & 31;
  int tileIdx = tid >> 8;
  int ntn     = N >> 4;
  int kt = tileIdx / ntn, nt = tileIdx % ntn;
  int n  = nt * 16 + (lane & 15);
  int k0 = kt * 32 + (v >> 2) * 16 + (lane >> 4) * 8 + 2 * (v & 3);
  Wp[tid] = pack2bf16(W[(size_t)k0 * N + n], W[(size_t)(k0 + 1) * N + n]);
}

// ---------------------------------------------------------------------------
// C[M x 256] = A[M x K] @ W  (+bias)(+add)(ReLU).  Block = 128 thr = 4 waves.
// Block owns one 16-row tile; wave w owns N-tiles 4w..4w+3.
// ---------------------------------------------------------------------------
__global__ __launch_bounds__(128) void k_gemm_n256(
    const float* __restrict__ A, const unsigned* __restrict__ Wp,
    float* __restrict__ C, const float* __restrict__ bias,
    const float* __restrict__ addv, int K, int doRelu) {
  __shared__ unsigned lA[16 * 132];          // max K=256: stride 132 dwords/row
  const int tid = threadIdx.x;
  const int rowBase = blockIdx.x * 16;
  const int strideDw = K / 2 + 4;            // pad: conflict-free ds_load_b128
  const int kp_per_row = K / 2;

  // cooperative f32 -> bf16 convert of the 16 x K A tile into LDS (row-major)
  for (int i = tid; i < 16 * kp_per_row; i += 128) {
    int r = i / kp_per_row, kp = i % kp_per_row;
    const float* s = A + (size_t)(rowBase + r) * K + 2 * kp;
    lA[r * strideDw + kp] = pack2bf16(s[0], s[1]);
  }
  __syncthreads();

  const int lane = tid & 31;
  const int wave = tid >> 5;
  const int lrow = lane & 15;                // A-frag source row (M)
  const int khalf = lane >> 4;
  const int jbase = wave * 4;                // 4 N-tiles per wave

  f32x8 acc[4] = {};
  for (int kt = 0; kt < K / 32; ++kt) {
    // A fragment (16x32 bf16): VGPR0-3 = K run1, VGPR4-7 = K run2 (+16)
    const unsigned* ap = &lA[lrow * strideDw + kt * 16 + khalf * 4];
    bf16x16 a;
    ((bf16x8*)&a)[0] = *(const bf16x8*)ap;
    ((bf16x8*)&a)[1] = *(const bf16x8*)(ap + 8);
#pragma unroll
    for (int j = 0; j < 4; ++j) {
      bf16x16 b = *(const bf16x16*)&Wp[(((size_t)kt * 16 + (jbase + j)) * 32 + lane) * 8];
      acc[j] = __builtin_amdgcn_wmma_f32_16x16x32_bf16(
          false, a, false, b, (short)0, acc[j], false, false);
    }
  }

  // D layout: lane l, vgpr v -> row = v + 8*(l>>4), col = (l&15)
#pragma unroll
  for (int j = 0; j < 4; ++j) {
    int n = (jbase + j) * 16 + lrow;
    float bv = bias ? bias[n] : 0.f;
#pragma unroll
    for (int v = 0; v < 8; ++v) {
      int row = rowBase + v + 8 * khalf;
      float val = acc[j][v] + bv;
      if (addv) val += addv[(size_t)row * 256 + n];
      if (doRelu) val = fmaxf(val, 0.f);
      C[(size_t)row * 256 + n] = val;
    }
  }
}

// --------------------------- degree / normalization -------------------------
__global__ void k_fill_deg(float* deg, int n) {
  int i = blockIdx.x * 256 + threadIdx.x;
  if (i < n) deg[i] = 1.0f;                  // self-loop
}
__global__ void k_count(const int* __restrict__ ei, float* deg, int n2e) {
  int i = blockIdx.x * 256 + threadIdx.x;
  if (i < n2e) atomAddF(&deg[ei[i]], 1.0f);
}
__global__ void k_dinv(float* deg, int n) {
  int i = blockIdx.x * 256 + threadIdx.x;
  if (i < n) deg[i] = rsqrtf(fmaxf(deg[i], 1.0f));
}

// x[i,:] = h[i,:] * dinv[i]^2   (self-loop message; also zero-inits the accum)
__global__ void k_selfloop(const float* __restrict__ h,
                           const float* __restrict__ dinv,
                           float* __restrict__ x) {
  size_t i = (size_t)blockIdx.x * 256 + threadIdx.x;   // grid = NNODES blocks
  float d = dinv[i >> 8];
  x[i] = h[i] * d * d;
}

// one wave per directed edge: x[d,:] += h[s,:] * dinv[s]*dinv[d]
__global__ __launch_bounds__(256) void k_scatter(
    const float* __restrict__ h, float* __restrict__ out,
    const float* __restrict__ dinv, const int* __restrict__ u,
    const int* __restrict__ p, int E) {
  int eid = blockIdx.x * 8 + (threadIdx.x >> 5);
  int lane = threadIdx.x & 31;
  int s, d;
  if (eid < E) { s = u[eid];     d = p[eid];     }
  else         { s = p[eid - E]; d = u[eid - E]; }
  float c = dinv[s] * dinv[d];
  const float* hp = h + (size_t)s * 256 + lane * 8;
  float*       op = out + (size_t)d * 256 + lane * 8;
  float4 a = *(const float4*)hp;
  float4 b = *(const float4*)(hp + 4);
  atomAddF(op + 0, a.x * c); atomAddF(op + 1, a.y * c);
  atomAddF(op + 2, a.z * c); atomAddF(op + 3, a.w * c);
  atomAddF(op + 4, b.x * c); atomAddF(op + 5, b.y * c);
  atomAddF(op + 6, b.z * c); atomAddF(op + 7, b.w * c);
}

__global__ void k_biasact(float* __restrict__ x, const float* __restrict__ b,
                          int doRelu) {
  size_t i = (size_t)blockIdx.x * 256 + threadIdx.x;
  float v = x[i] + b[threadIdx.x];           // 256 cols == blockDim
  x[i] = doRelu ? fmaxf(v, 0.f) : v;
}

// ---------------------------------------------------------------------------
// Fused edge MLP: per 16-edge tile, gather pair=[user_out|prod_out] (16x512)
// into LDS as bf16; h = relu(pair@m1W + m1b) via 16x16 WMMAs kept in-register;
// preds = h @ m2W + m2b folded into the accumulator walk. One wave per tile.
// ---------------------------------------------------------------------------
__global__ __launch_bounds__(32) void k_edge_mlp(
    const float* __restrict__ x, const int* __restrict__ uidx,
    const int* __restrict__ pidx, const unsigned* __restrict__ m1Wp,
    const float* __restrict__ m1b, const float* __restrict__ m2W,
    const float* __restrict__ m2b, float* __restrict__ out) {
  __shared__ unsigned lA[16 * 260];          // 16 rows x 512 bf16, padded
  const int tile = blockIdx.x;
  const int lane = threadIdx.x;

  for (int e = 0; e < 16; ++e) {
    int ui = uidx[tile * 16 + e];
    int pi = pidx[tile * 16 + e];            // already global node index
    const float* su = x + (size_t)ui * 256;
    const float* sp = x + (size_t)pi * 256;
#pragma unroll
    for (int it = 0; it < 2; ++it) {
      int k = it * 128 + lane * 4;
      float4 vu = *(const float4*)(su + k);
      lA[e * 260 + k / 2]           = pack2bf16(vu.x, vu.y);
      lA[e * 260 + k / 2 + 1]       = pack2bf16(vu.z, vu.w);
      float4 vp = *(const float4*)(sp + k);
      lA[e * 260 + 128 + k / 2]     = pack2bf16(vp.x, vp.y);
      lA[e * 260 + 128 + k / 2 + 1] = pack2bf16(vp.z, vp.w);
    }
  }
  __syncthreads();

  const int lrow = lane & 15, khalf = lane >> 4;
  const float m2b0 = m2b[0];
  float parts[8] = {0.f, 0.f, 0.f, 0.f, 0.f, 0.f, 0.f, 0.f};

  for (int jn = 0; jn < 16; ++jn) {
    float b1 = m1b[jn * 16 + lrow];
    float w2 = m2W[jn * 16 + lrow];
    f32x8 acc = {};
#pragma unroll
    for (int kt = 0; kt < 16; ++kt) {
      const unsigned* ap = &lA[lrow * 260 + kt * 16 + khalf * 4];
      bf16x16 a;
      ((bf16x8*)&a)[0] = *(const bf16x8*)ap;
      ((bf16x8*)&a)[1] = *(const bf16x8*)(ap + 8);
      bf16x16 b = *(const bf16x16*)&m1Wp[(((size_t)kt * 16 + jn) * 32 + lane) * 8];
      acc = __builtin_amdgcn_wmma_f32_16x16x32_bf16(
          false, a, false, b, (short)0, acc, false, false);
    }
#pragma unroll
    for (int v = 0; v < 8; ++v)
      parts[v] += fmaxf(acc[v] + b1, 0.f) * w2;
  }

  // reduce over the 16 lanes of each half (xor of bits 0..3 stays in-half)
#pragma unroll
  for (int m = 8; m >= 1; m >>= 1)
#pragma unroll
    for (int v = 0; v < 8; ++v)
      parts[v] += __shfl_xor(parts[v], m);

  if ((lane & 15) == 0) {
    int ebase = tile * 16 + khalf * 8;       // lanes 16-31 hold rows 8..15
#pragma unroll
    for (int v = 0; v < 8; ++v)
      out[ebase + v] = parts[v] + m2b0;
  }
}

// ---------------------------------------------------------------------------
extern "C" void kernel_launch(void* const* d_in, const int* in_sizes, int n_in,
                              void* d_out, int out_size, void* d_ws, size_t ws_size,
                              hipStream_t stream) {
  (void)n_in; (void)out_size; (void)ws_size;
  const int*   edge      = (const int*)d_in[0];
  const float* user_feat = (const float*)d_in[1];
  const float* prod_feat = (const float*)d_in[2];
  const float* user_emb  = (const float*)d_in[3];
  const float* prod_emb  = (const float*)d_in[4];
  const float* uW  = (const float*)d_in[5];
  const float* ub  = (const float*)d_in[6];
  const float* pW  = (const float*)d_in[7];
  const float* pb  = (const float*)d_in[8];
  const float* c1W = (const float*)d_in[9];
  const float* c1b = (const float*)d_in[10];
  const float* c2W = (const float*)d_in[11];
  const float* c2b = (const float*)d_in[12];
  const float* m1W = (const float*)d_in[13];
  const float* m1b = (const float*)d_in[14];
  const float* m2W = (const float*)d_in[15];
  const float* m2b = (const float*)d_in[16];
  float* preds = (float*)d_out;

  const int E = in_sizes[0] / 2;
  const int* uidx = edge;
  const int* pidx = edge + E;

  // workspace carve-up (~309 MB): x, h node buffers, dinv, packed weights
  float* ws = (float*)d_ws;
  size_t off = 0;
  float* x    = ws + off; off += (size_t)NNODES * 256;
  float* h    = ws + off; off += (size_t)NNODES * 256;
  float* dinv = ws + off; off += 150016;                    // deg -> dinv in place
  unsigned* uWp  = (unsigned*)(ws + off); off += 128 * 256 / 2;
  unsigned* pWp  = (unsigned*)(ws + off); off += 128 * 256 / 2;
  unsigned* c1Wp = (unsigned*)(ws + off); off += 256 * 256 / 2;
  unsigned* c2Wp = (unsigned*)(ws + off); off += 256 * 256 / 2;
  unsigned* m1Wp = (unsigned*)(ws + off); off += 512 * 256 / 2;

  // 1) pack all weights to bf16 WMMA B-fragments (weights stay L2-resident)
  k_pack_w<<<(128 * 256 / 2 + 255) / 256, 256, 0, stream>>>(uW,  uWp,  128, 256);
  k_pack_w<<<(128 * 256 / 2 + 255) / 256, 256, 0, stream>>>(pW,  pWp,  128, 256);
  k_pack_w<<<(256 * 256 / 2 + 255) / 256, 256, 0, stream>>>(c1W, c1Wp, 256, 256);
  k_pack_w<<<(256 * 256 / 2 + 255) / 256, 256, 0, stream>>>(c2W, c2Wp, 256, 256);
  k_pack_w<<<(512 * 256 / 2 + 255) / 256, 256, 0, stream>>>(m1W, m1Wp, 512, 256);

  // 2) node feature transforms: x = feat@W + b + emb
  k_gemm_n256<<<NUSERS / 16, 128, 0, stream>>>(user_feat, uWp, x, ub, user_emb, 128, 0);
  k_gemm_n256<<<NPRODS / 16, 128, 0, stream>>>(prod_feat, pWp, x + (size_t)NUSERS * 256,
                                               pb, prod_emb, 128, 0);

  // 3) symmetric-norm degree factors
  k_fill_deg<<<(NNODES + 255) / 256, 256, 0, stream>>>(dinv, NNODES);
  k_count<<<(2 * E + 255) / 256, 256, 0, stream>>>(edge, dinv, 2 * E);
  k_dinv<<<(NNODES + 255) / 256, 256, 0, stream>>>(dinv, NNODES);

  // 4) conv1: h = x@W1 ; x = relu(scatter(h) + selfloop + b1)
  k_gemm_n256<<<NNODES / 16, 128, 0, stream>>>(x, c1Wp, h, nullptr, nullptr, 256, 0);
  k_selfloop<<<NNODES, 256, 0, stream>>>(h, dinv, x);
  k_scatter<<<(2 * E) / 8, 256, 0, stream>>>(h, x, dinv, uidx, pidx, E);
  k_biasact<<<NNODES, 256, 0, stream>>>(x, c1b, 1);

  // 5) conv2 (no relu)
  k_gemm_n256<<<NNODES / 16, 128, 0, stream>>>(x, c2Wp, h, nullptr, nullptr, 256, 0);
  k_selfloop<<<NNODES, 256, 0, stream>>>(h, dinv, x);
  k_scatter<<<(2 * E) / 8, 256, 0, stream>>>(h, x, dinv, uidx, pidx, E);
  k_biasact<<<NNODES, 256, 0, stream>>>(x, c2b, 0);

  // 6) fused edge MLP -> preds (hidden matrix never hits memory)
  k_edge_mlp<<<E / 16, 32, 0, stream>>>(x, uidx, pidx, m1Wp, m1b, m2W, m2b, preds);
}
